// GRULayer_20461224198664
// MI455X (gfx1250) — matrix operand order repaired and checked by reading, not compiled, not run on hardware
//
#include <hip/hip_runtime.h>

// ---------------------------------------------------------------------------
// GRU-style scan for MI455X (gfx1250, wave32, WMMA bf16 path)
//   T=512 steps, B=128 batch, D=256 in-dim, H=256 hidden
// Design:
//   1) cvt_weights : W/U fp32 -> bf16, transposed to [gate][n][k]
//   2) proj        : x{u,r,c} = h_t @ W + b  via v_wmma_f32_16x16x32_bf16.
//                    Weights staged to LDS with global_load_async_to_lds_b128,
//                    B-fragments come from bank-conflict-free ds_load_b128.
//   3) scan        : persistent per-batch-slab recurrence.
//                    - Uu+Ur (256KB bf16) async-staged into the 320KB WGP LDS
//                    - Uc B-fragments pinned in VGPRs (64 regs/lane)
//                    - x projections double-buffered in registers, loads issued
//                      inside the split-barrier window (signal .. wait)
//                    -> per-step critical path touches only LDS + WMMA
// ---------------------------------------------------------------------------

#define TT 512
#define BB 128
#define DD 256
#define HH 256
#define PAD 8                       // row padding (elems) -> 528B stride, 4-bank skew
#define US (DD + PAD)               // padded k-stride for staged weight rows
#define CS (HH + PAD)               // padded k-stride for c-state rows

typedef __attribute__((ext_vector_type(16))) __bf16 v16bf;
typedef __attribute__((ext_vector_type(8)))  __bf16 v8bf;
typedef __attribute__((ext_vector_type(8)))  float  v8f;
typedef __attribute__((ext_vector_type(4)))  float  v4f;

union BF16x16 { v16bf v; v8bf h[2]; };

__device__ __forceinline__ float fast_sigmoid(float x) {
    return 1.0f / (1.0f + __expf(-x));
}
__device__ __forceinline__ float fast_tanh(float x) {
    float e = __expf(2.0f * x);
    return (e - 1.0f) / (e + 1.0f);
}

// ---- CDNA5 async copy: global -> LDS, 16B per lane, tracked by ASYNCcnt ----
__device__ __forceinline__ void async_copy_b128(void* lds, const void* gptr) {
    unsigned ldsOff = (unsigned)(size_t)lds;    // LDS aperture: low 32 bits = LDS byte addr
    asm volatile("global_load_async_to_lds_b128 %0, %1, off"
                 :: "v"(ldsOff), "v"(gptr) : "memory");
}
__device__ __forceinline__ void wait_async0() {
    asm volatile("s_wait_asynccnt 0" ::: "memory");
}
__device__ __forceinline__ void wait_ds0() {
    asm volatile("s_wait_dscnt 0" ::: "memory");
}

// ---- split workgroup barrier (signal ... <independent work> ... wait) ------
#if __has_builtin(__builtin_amdgcn_s_barrier_signal) && __has_builtin(__builtin_amdgcn_s_barrier_wait)
#define BAR_SIGNAL() __builtin_amdgcn_s_barrier_signal(-1)
#define BAR_WAIT()   __builtin_amdgcn_s_barrier_wait(-1)
#else
#define BAR_SIGNAL() ((void)0)
#define BAR_WAIT()   __syncthreads()
#endif

// ---------------------------------------------------------------------------
// Kernel 1: convert + transpose weights to bf16.
// Wt[g][n][k] = W_g[k][n],  Ut[g][n][k] = U_g[k][n];  g: 0=u, 1=r, 2=c
// ---------------------------------------------------------------------------
__global__ void cvt_weights(const float* __restrict__ Wu, const float* __restrict__ Wr,
                            const float* __restrict__ Wc, const float* __restrict__ Uu,
                            const float* __restrict__ Ur, const float* __restrict__ Uc,
                            __bf16* __restrict__ Wt, __bf16* __restrict__ Ut) {
    int idx = blockIdx.x * blockDim.x + threadIdx.x;      // < 3*H*D
    int g = idx / (HH * DD);
    int r = idx - g * (HH * DD);
    int n = r / DD;
    int k = r - n * DD;
    const float* W = (g == 0) ? Wu : ((g == 1) ? Wr : Wc);
    const float* U = (g == 0) ? Uu : ((g == 1) ? Ur : Uc);
    Wt[idx] = (__bf16)W[(size_t)k * HH + n];
    Ut[idx] = (__bf16)U[(size_t)k * HH + n];
}

// ---------------------------------------------------------------------------
// Kernel 2: input projection GEMM.
//   grid = (T, 3 gates, 2 N-halves), block = 256 (8 waves).
//   Each WG computes a [B=128, 128] tile; wave w -> rows 16w..16w+15,
//   8 N-tiles per wave (64 accum VGPRs, no spills).
//   Weight half-slab (64KB) async-staged to LDS with padded rows.
// ---------------------------------------------------------------------------
__global__ void __launch_bounds__(256, 2)
proj(const float* __restrict__ h_t, const __bf16* __restrict__ Wt,
     const float* __restrict__ bu, const float* __restrict__ br,
     const float* __restrict__ bc, __bf16* __restrict__ X) {
    const int t     = blockIdx.x;
    const int g     = blockIdx.y;
    const int nhalf = blockIdx.z;
    const int tid   = threadIdx.x;
    const int wave  = tid >> 5;
    const int lane  = tid & 31;
    const int ml    = lane & 15;       // fragment row / col index
    const int hi    = lane >> 4;       // half-wave select

    extern __shared__ __align__(16) char smem_raw[];
    __bf16* Wlds = (__bf16*)smem_raw;                 // [128 rows][US] padded

    const __bf16* Wsrc = Wt + ((size_t)g * HH + nhalf * 128) * DD;
    __bf16*       Xg   = X  + (size_t)g * TT * HH * BB;

    // ---- async-stage 128 rows x 256 k of bf16 weights into LDS (padded rows)
    for (int c = tid; c < 128 * 32; c += 256) {
        const int row = c >> 5, ch = c & 31;
        async_copy_b128(Wlds + (size_t)row * US + ch * 8, Wsrc + (size_t)row * DD + ch * 8);
    }
    wait_async0();
    __syncthreads();

    v8f acc[8];
    #pragma unroll
    for (int nt = 0; nt < 8; ++nt) acc[nt] = (v8f)(0.0f);

    const int bb = wave * 16 + ml;                    // batch row feeding A
    const float* arowbase = h_t + ((size_t)t * BB + bb) * DD;

    for (int kk = 0; kk < 8; ++kk) {
        // A fragment: 16x32 bf16 (ISA layout: half-waves split k by +8/+16)
        const float* ar = arowbase + kk * 32 + hi * 8;
        float va[16];
        *(v4f*)&va[0]  = *(const v4f*)(ar + 0);
        *(v4f*)&va[4]  = *(const v4f*)(ar + 4);
        *(v4f*)&va[8]  = *(const v4f*)(ar + 16);
        *(v4f*)&va[12] = *(const v4f*)(ar + 20);
        v16bf afrag;
        #pragma unroll
        for (int i = 0; i < 16; ++i) afrag[i] = (__bf16)va[i];

        #pragma unroll
        for (int nt = 0; nt < 8; ++nt) {
            const int nl = nt * 16 + ml;              // local column
            const __bf16* bp = Wlds + (size_t)nl * US + kk * 32 + hi * 8;
            BF16x16 bfrag;
            bfrag.h[0] = *(const v8bf*)(bp);
            bfrag.h[1] = *(const v8bf*)(bp + 16);
            acc[nt] = __builtin_amdgcn_wmma_f32_16x16x32_bf16(
                false, afrag, false, bfrag.v, (short)0, acc[nt], false, false);
        }
    }

    // ---- bias + store bf16 to X[t][n][b]
    const float* bias = (g == 0) ? bu : ((g == 1) ? br : bc);
    #pragma unroll
    for (int nt = 0; nt < 8; ++nt) {
        const int n  = nhalf * 128 + nt * 16 + ml;
        const float bv = bias[n];
        #pragma unroll
        for (int r = 0; r < 8; ++r) {
            const int brow = wave * 16 + r + 8 * hi;  // batch index of elem r
            Xg[((size_t)t * HH + n) * BB + brow] = (__bf16)(acc[nt][r] + bv);
        }
    }
}

// ---------------------------------------------------------------------------
// Kernel 3: persistent recurrence. grid = 8 WGs (batch slabs of 16 rows),
// block = 512 (16 waves). Wave w owns output columns [16w, 16w+16).
// LDS: cA(16xCS) | aA(16xCS) | Uu,Ur staged (2 x 256 x US) ~= 281KB.
// Uc fragments live in VGPRs; x projections double-buffered in VGPRs.
// ---------------------------------------------------------------------------
__global__ void __launch_bounds__(512, 1)
scan(const float* __restrict__ init, const __bf16* __restrict__ Ut,
     const __bf16* __restrict__ X, float* __restrict__ out) {
    const int b0   = blockIdx.x * 16;
    const int tid  = threadIdx.x;
    const int wave = tid >> 5;
    const int lane = tid & 31;
    const int ml   = lane & 15;
    const int hi   = lane >> 4;
    const int n    = wave * 16 + ml;          // output column this lane owns

    extern __shared__ __align__(16) char smem_raw[];
    __bf16* cA   = (__bf16*)smem_raw;                 // [16][CS]
    __bf16* aA   = cA + 16 * CS;                      // [16][CS]
    __bf16* Ulds = aA + 16 * CS;                      // [2*256][US]  (Uu rows, then Ur rows)

    // ---- async-stage Uu + Ur (contiguous 256KB source) into padded LDS rows
    for (int c = tid; c < 512 * 32; c += 512) {
        const int row = c >> 5, ch = c & 31;
        async_copy_b128(Ulds + (size_t)row * US + ch * 8, Ut + (size_t)row * DD + ch * 8);
    }
    wait_async0();

    // seed state from init
    for (int i = tid; i < 16 * HH; i += 512) {
        const int row = i >> 8, col = i & 255;
        cA[row * CS + col] = (__bf16)init[(size_t)(b0 + row) * HH + col];
    }

    const __bf16* UuL = Ulds;                         // LDS-resident
    const __bf16* UrL = Ulds + (size_t)HH * US;
    const __bf16* UcT = Ut + (size_t)2 * HH * DD;     // source for register-pinned frags
    const __bf16* Xu  = X;
    const __bf16* Xr  = X + (size_t)TT * HH * BB;
    const __bf16* Xc  = X + (size_t)2 * TT * HH * BB;

    // ---- pin this wave's Uc B-fragments in VGPRs for the whole scan ----
    BF16x16 ucf[8];
    #pragma unroll
    for (int kk = 0; kk < 8; ++kk) {
        const size_t bo = (size_t)n * DD + kk * 32 + hi * 8;
        ucf[kk].h[0] = *(const v8bf*)(UcT + bo);
        ucf[kk].h[1] = *(const v8bf*)(UcT + bo + 16);
    }

    // ---- prime the x double-buffer for t = 0 ----
    size_t xoff = ((size_t)0 * HH + n) * BB + b0 + hi * 8;
    v8bf xu_c = *(const v8bf*)(Xu + xoff);
    v8bf xr_c = *(const v8bf*)(Xr + xoff);
    v8bf xc_c = *(const v8bf*)(Xc + xoff);

    __syncthreads();

    for (int t = 0; t < TT; ++t) {
        // ---------------- phase 1: update & reset gates -----------------
        v8f accU = (v8f)(0.0f);
        v8f accR = (v8f)(0.0f);
        #pragma unroll
        for (int kk = 0; kk < 8; ++kk) {
            const __bf16* ap = cA + ml * CS + kk * 32 + hi * 8;
            BF16x16 afrag;
            afrag.h[0] = *(const v8bf*)(ap);
            afrag.h[1] = *(const v8bf*)(ap + 16);

            const size_t bo = (size_t)n * US + kk * 32 + hi * 8;
            BF16x16 bU, bR;
            bU.h[0] = *(const v8bf*)(UuL + bo);
            bU.h[1] = *(const v8bf*)(UuL + bo + 16);
            bR.h[0] = *(const v8bf*)(UrL + bo);
            bR.h[1] = *(const v8bf*)(UrL + bo + 16);

            accU = __builtin_amdgcn_wmma_f32_16x16x32_bf16(
                false, afrag.v, false, bU.v, (short)0, accU, false, false);
            accR = __builtin_amdgcn_wmma_f32_16x16x32_bf16(
                false, afrag.v, false, bR.v, (short)0, accR, false, false);
        }

        float gu[8], co[8];
        #pragma unroll
        for (int r = 0; r < 8; ++r) {
            const int m = r + 8 * hi;
            gu[r] = fast_sigmoid((float)xu_c[r] + accU[r]);
            const float gr = fast_sigmoid((float)xr_c[r] + accR[r]);
            co[r] = (float)cA[m * CS + n];
            aA[m * CS + n] = (__bf16)(gr * co[r]);
        }
        wait_ds0();
        BAR_SIGNAL();

        // ---- overlap barrier wait with next-step x loads (global/L2) ----
        const int tn = (t + 1 < TT) ? t + 1 : t;
        const size_t xoff_n = ((size_t)tn * HH + n) * BB + b0 + hi * 8;
        v8bf xu_n = *(const v8bf*)(Xu + xoff_n);
        v8bf xr_n = *(const v8bf*)(Xr + xoff_n);
        v8bf xc_n = *(const v8bf*)(Xc + xoff_n);

        BAR_WAIT();

        // ---------------- phase 2: candidate + blend --------------------
        v8f accC = (v8f)(0.0f);
        #pragma unroll
        for (int kk = 0; kk < 8; ++kk) {
            const __bf16* ap = aA + ml * CS + kk * 32 + hi * 8;
            BF16x16 afrag;
            afrag.h[0] = *(const v8bf*)(ap);
            afrag.h[1] = *(const v8bf*)(ap + 16);

            accC = __builtin_amdgcn_wmma_f32_16x16x32_bf16(
                false, afrag.v, false, ucf[kk].v, (short)0, accC, false, false);
        }

        #pragma unroll
        for (int r = 0; r < 8; ++r) {
            const int m  = r + 8 * hi;
            const float ct = fast_tanh((float)xc_c[r] + accC[r]);
            const float cn = (1.0f - gu[r]) * co[r] + gu[r] * ct;
            out[((size_t)t * BB + b0 + m) * HH + n] = cn;
            cA[m * CS + n] = (__bf16)cn;
        }
        wait_ds0();
        BAR_SIGNAL();
        xu_c = xu_n;                      // rotate double-buffer during barrier
        xr_c = xr_n;
        xc_c = xc_n;
        BAR_WAIT();
    }
}

// ---------------------------------------------------------------------------
// Host launcher
// ---------------------------------------------------------------------------
extern "C" void kernel_launch(void* const* d_in, const int* in_sizes, int n_in,
                              void* d_out, int out_size, void* d_ws, size_t ws_size,
                              hipStream_t stream) {
    // setup_inputs order: h_t, init, Wc, Wu, Wr, Uc, Uu, Ur, bc, bu, br
    const float* h_t  = (const float*)d_in[0];
    const float* init = (const float*)d_in[1];
    const float* Wc   = (const float*)d_in[2];
    const float* Wu   = (const float*)d_in[3];
    const float* Wr   = (const float*)d_in[4];
    const float* Uc   = (const float*)d_in[5];
    const float* Uu   = (const float*)d_in[6];
    const float* Ur   = (const float*)d_in[7];
    const float* bc   = (const float*)d_in[8];
    const float* bu   = (const float*)d_in[9];
    const float* br   = (const float*)d_in[10];
    float* out = (float*)d_out;

    // workspace layout (bf16): X[3][T][H][B] (~96MB), Wt[3][H][D], Ut[3][H][D]
    __bf16* X  = (__bf16*)d_ws;
    __bf16* Wt = X  + (size_t)3 * TT * HH * BB;
    __bf16* Ut = Wt + (size_t)3 * HH * DD;

    const size_t projLds = (size_t)128 * US * sizeof(__bf16);                    // ~66KB
    const size_t scanLds = (size_t)(2 * 16 * CS + 2 * HH * US) * sizeof(__bf16); // ~281KB

    cvt_weights<<<(3 * HH * DD) / 256, 256, 0, stream>>>(Wu, Wr, Wc, Uu, Ur, Uc, Wt, Ut);
    proj<<<dim3(TT, 3, 2), 256, projLds, stream>>>(h_t, Wt, bu, br, bc, X);
    scan<<<BB / 16, 512, scanLds, stream>>>(init, Ut, X, out);
}
